// Decoder_50294067036730
// MI455X (gfx1250) — matrix operand order repaired and checked by reading, not compiled, not run on hardware
//
#include <hip/hip_runtime.h>
#include <hip/hip_bf16.h>

typedef __attribute__((ext_vector_type(16))) __bf16 v16bf;
typedef __attribute__((ext_vector_type(8)))  float  v8f;

#define CV    50000
#define CVG   20000
#define CE    128
#define CH    256
#define CTP   400
#define CB    32
#define CTDEC 32
#define KLSTM (CE + CH)   /* 384 */
#define GATES (4 * CH)    /* 1024 */

__device__ __forceinline__ __bf16 f2bf(float f) {
    unsigned u = __float_as_uint(f);
    unsigned r = (u + 0x7FFFu + ((u >> 16) & 1u)) >> 16;
    unsigned short s = (unsigned short)r;
    __bf16 o;
    __builtin_memcpy(&o, &s, 2);
    return o;
}

// ---------------- prep kernels ----------------

__global__ void k_cvt_wgen(const float* __restrict__ wg, __bf16* __restrict__ wgb) {
    int i = blockIdx.x * blockDim.x + threadIdx.x;
    if (i < CVG * CH) wgb[i] = f2bf(wg[i]);
}

__global__ void k_build_wx(const float* __restrict__ W_ih, const float* __restrict__ W_hh,
                           __bf16* __restrict__ wxb) {
    int i = blockIdx.x * blockDim.x + threadIdx.x;
    if (i >= GATES * KLSTM) return;
    int row = i / KLSTM, k = i % KLSTM;
    float v = (k < CE) ? W_ih[row * CE + k] : W_hh[row * CH + (k - CE)];
    wxb[i] = f2bf(v);
}

// ctx_proj[t,b,g] = sum_h context[t,b,h] * W_h[g,h]
__global__ void k_ctxproj(const float* __restrict__ ctx, const float* __restrict__ W_h,
                          float* __restrict__ ctx_proj) {
    int tb = blockIdx.x;           // t*B + b
    int g  = threadIdx.x;          // 0..255
    __shared__ float row[CH];
    row[g] = ctx[tb * CH + g];
    __syncthreads();
    const float* wr = W_h + g * CH;
    float acc = 0.f;
#pragma unroll 4
    for (int k = 0; k < CH; ++k) acc += row[k] * wr[k];
    ctx_proj[tb * CH + g] = acc;
}

__global__ void k_init(const float* __restrict__ enc_hidden, const int* __restrict__ a_batch,
                       float* __restrict__ h, float* __restrict__ c,
                       float* __restrict__ cov, int* __restrict__ tok) {
    int i = blockIdx.x * blockDim.x + threadIdx.x;
    if (i < CB * CH) { h[i] = enc_hidden[i]; c[i] = 0.f; }
    if (i < CB * CTP) cov[i] = 0.f;
    if (i < CB) tok[i] = a_batch[i];   // a_batch[0][b]
}

// ---------------- per-step kernels ----------------

// gather embedding + stage bf16 X = [emb | h]
__global__ void k_stage_x(const float* __restrict__ word_lut, const int* __restrict__ tok,
                          const float* __restrict__ h, float* __restrict__ emb,
                          __bf16* __restrict__ x_bf) {
    int b = blockIdx.x, i = threadIdx.x;   // blockDim = 384
    if (i < CE) {
        float e = word_lut[(long)tok[b] * CE + i];
        emb[b * CE + i] = e;
        x_bf[b * KLSTM + i] = f2bf(e);
    } else {
        x_bf[b * KLSTM + i] = f2bf(h[b * CH + (i - CE)]);
    }
}

// A fragment loader: contiguous K order, same as B (lane<16: K 0..15; lane>=16: K 16..31)
__device__ __forceinline__ v16bf load_a(const __bf16* __restrict__ ar, int k0, int hf) {
    v16bf a;
#pragma unroll
    for (int e = 0; e < 16; ++e) {
        a[e] = ar[k0 + hf * 16 + e];
    }
    return a;
}

// gates = X @ Wx^T + b_ih + b_hh   (WMMA bf16, M=32,N=1024,K=384; both M-tiles per wave)
__global__ void k_gates_wmma(const __bf16* __restrict__ x_bf, const __bf16* __restrict__ wxb,
                             const float* __restrict__ b_ih, const float* __restrict__ b_hh,
                             float* __restrict__ gates) {
    int wid = blockIdx.x * (blockDim.x >> 5) + (threadIdx.x >> 5);
    const int NT = GATES / 16;                 // 64 N-tiles
    if (wid >= NT) return;                     // wave-uniform
    int nt = wid;
    int lane = threadIdx.x & 31;
    int hf = lane >> 4, l16 = lane & 15;
    int ncol = nt * 16 + l16;
    const __bf16* a0r = x_bf + l16 * KLSTM;          // rows 0..15
    const __bf16* a1r = x_bf + (16 + l16) * KLSTM;   // rows 16..31
    const __bf16* br  = wxb + (long)ncol * KLSTM;
    float bias = b_ih[ncol] + b_hh[ncol];
    v8f acc0, acc1;
#pragma unroll
    for (int r = 0; r < 8; ++r) { acc0[r] = bias; acc1[r] = bias; }
#pragma unroll
    for (int kk = 0; kk < KLSTM / 32; ++kk) {
        int k0 = kk * 32;
        v16bf bb;
#pragma unroll
        for (int e = 0; e < 16; ++e) bb[e] = br[k0 + hf * 16 + e];
        v16bf a0 = load_a(a0r, k0, hf);
        v16bf a1 = load_a(a1r, k0, hf);
        acc0 = __builtin_amdgcn_wmma_f32_16x16x32_bf16(false, a0, false, bb,
                                                       (short)0, acc0, false, false);
        acc1 = __builtin_amdgcn_wmma_f32_16x16x32_bf16(false, a1, false, bb,
                                                       (short)0, acc1, false, false);
    }
#pragma unroll
    for (int r = 0; r < 8; ++r) {
        int m = hf * 8 + r;
        gates[m * GATES + ncol]        = acc0[r];
        gates[(16 + m) * GATES + ncol] = acc1[r];
    }
}

__global__ void k_lstm(const float* __restrict__ gates, float* __restrict__ h,
                       float* __restrict__ c) {
    int b = blockIdx.x, j = threadIdx.x;
    const float* g = gates + b * GATES;
    float ig = g[j], fg = g[CH + j], gg = g[2 * CH + j], og = g[3 * CH + j];
    float si = 1.f / (1.f + __expf(-ig));
    float sf = 1.f / (1.f + __expf(-fg));
    float so = 1.f / (1.f + __expf(-og));
    float cn = sf * c[b * CH + j] + si * tanhf(gg);
    c[b * CH + j] = cn;
    h[b * CH + j] = so * tanhf(cn);
}

// sproj = h @ W_s^T
__global__ void k_sproj(const float* __restrict__ h, const float* __restrict__ W_s,
                        float* __restrict__ sproj) {
    int b = blockIdx.x, g = threadIdx.x;
    __shared__ float hr[CH];
    hr[g] = h[b * CH + g];
    __syncthreads();
    const float* wr = W_s + g * CH;
    float acc = 0.f;
#pragma unroll 4
    for (int k = 0; k < CH; ++k) acc += hr[k] * wr[k];
    sproj[b * CH + g] = acc;
}

// scores[b][t] = v_attn . tanh(ctx_proj + sproj + cov*w_c + b_attn)
__global__ void k_score(const float* __restrict__ ctx_proj, const float* __restrict__ sproj,
                        const float* __restrict__ cov, const float* __restrict__ w_c,
                        const float* __restrict__ b_attn, const float* __restrict__ v_attn,
                        float* __restrict__ scores) {
    int t = blockIdx.x, b = blockIdx.y, i = threadIdx.x;
    float f = tanhf(ctx_proj[(t * CB + b) * CH + i] + sproj[b * CH + i] +
                    cov[b * CTP + t] * w_c[i] + b_attn[i]);
    __shared__ float sh[CH];
    sh[i] = f * v_attn[i];
    __syncthreads();
    for (int s = CH / 2; s > 0; s >>= 1) {
        if (i < s) sh[i] += sh[i + s];
        __syncthreads();
    }
    if (i == 0) scores[b * CTP + t] = sh[0];
}

__global__ void k_softmax_attn(const float* __restrict__ scores, float* __restrict__ attn,
                               float* __restrict__ cov) {
    int b = blockIdx.x, i = threadIdx.x;   // blockDim = 512
    __shared__ float sh[512];
    float s = (i < CTP) ? scores[b * CTP + i] : -1e30f;
    sh[i] = s;
    __syncthreads();
    for (int st = 256; st > 0; st >>= 1) {
        if (i < st) sh[i] = fmaxf(sh[i], sh[i + st]);
        __syncthreads();
    }
    float m = sh[0];
    __syncthreads();
    float e = (i < CTP) ? __expf(s - m) : 0.f;
    sh[i] = e;
    __syncthreads();
    for (int st = 256; st > 0; st >>= 1) {
        if (i < st) sh[i] += sh[i + st];
        __syncthreads();
    }
    float inv = 1.f / sh[0];
    if (i < CTP) {
        float a = e * inv;
        attn[b * CTP + i] = a;
        cov[b * CTP + i] += a;   // coverage for next step
    }
}

__global__ void k_ctxvec(const float* __restrict__ attn, const float* __restrict__ context,
                         float* __restrict__ ctxvec) {
    int b = blockIdx.x, i = threadIdx.x;
    __shared__ float a[CTP];
    for (int t = i; t < CTP; t += blockDim.x) a[t] = attn[b * CTP + t];
    __syncthreads();
    float acc = 0.f;
    for (int t = 0; t < CTP; ++t) acc += a[t] * context[(t * CB + b) * CH + i];
    ctxvec[b * CH + i] = acc;
}

__global__ void k_attnh(const float* __restrict__ ctxvec, const float* __restrict__ h,
                        const float* __restrict__ W_out, const float* __restrict__ b_out,
                        float* __restrict__ attnh, __bf16* __restrict__ attnh_bf) {
    int b = blockIdx.x, g = threadIdx.x;
    __shared__ float xin[2 * CH];
    xin[g] = ctxvec[b * CH + g];
    xin[CH + g] = h[b * CH + g];
    __syncthreads();
    const float* wr = W_out + g * 2 * CH;
    float acc = b_out[g];
#pragma unroll 4
    for (int k = 0; k < 2 * CH; ++k) acc += xin[k] * wr[k];
    float v = tanhf(acc);
    attnh[b * CH + g] = v;
    attnh_bf[b * CH + g] = f2bf(v);
}

__global__ void k_pgen(const float* __restrict__ attnh, const float* __restrict__ emb,
                       const float* __restrict__ w_p1, const float* __restrict__ w_p2,
                       float* __restrict__ pgen, float* __restrict__ trace, int step) {
    int b = blockIdx.x, i = threadIdx.x;
    float v = attnh[b * CH + i] * w_p1[i];
    if (i < CE) v += emb[b * CE + i] * w_p2[i];
    __shared__ float sh[CH];
    sh[i] = v;
    __syncthreads();
    for (int s = CH / 2; s > 0; s >>= 1) {
        if (i < s) sh[i] += sh[i + s];
        __syncthreads();
    }
    if (i == 0) {
        float p = 1.f / (1.f + __expf(-sh[0]));
        pgen[b] = p;
        if (b == 0) trace[step] = p;   // p_gens[:,0]
    }
}

// logits = attn_h @ W_gen^T + b_gen (WMMA bf16, M=32,N=20000,K=256; both M-tiles per
// wave so each W_gen fragment is loaded once and reused by two WMMAs)
__global__ void k_gen_wmma(const __bf16* __restrict__ attnh_bf, const __bf16* __restrict__ wgb,
                           const float* __restrict__ b_gen, float* __restrict__ gen) {
    int wid = blockIdx.x * (blockDim.x >> 5) + (threadIdx.x >> 5);
    const int NT = CVG / 16;                 // 1250 N-tiles
    if (wid >= NT) return;                   // wave-uniform tail exit
    int nt = wid;
    int lane = threadIdx.x & 31;
    int hf = lane >> 4, l16 = lane & 15;
    int ncol = nt * 16 + l16;
    const __bf16* a0r = attnh_bf + l16 * CH;          // rows 0..15
    const __bf16* a1r = attnh_bf + (16 + l16) * CH;   // rows 16..31
    const __bf16* br  = wgb + (long)ncol * CH;
    float bias = b_gen[ncol];
    v8f acc0, acc1;
#pragma unroll
    for (int r = 0; r < 8; ++r) { acc0[r] = bias; acc1[r] = bias; }
#pragma unroll
    for (int kk = 0; kk < CH / 32; ++kk) {
        int k0 = kk * 32;
        __builtin_prefetch(br + k0 + 64, 0, 1);   // global_prefetch_b8 ahead in K
        v16bf bb;
#pragma unroll
        for (int e = 0; e < 16; ++e) bb[e] = br[k0 + hf * 16 + e];
        v16bf a0 = load_a(a0r, k0, hf);
        v16bf a1 = load_a(a1r, k0, hf);
        acc0 = __builtin_amdgcn_wmma_f32_16x16x32_bf16(false, a0, false, bb,
                                                       (short)0, acc0, false, false);
        acc1 = __builtin_amdgcn_wmma_f32_16x16x32_bf16(false, a1, false, bb,
                                                       (short)0, acc1, false, false);
    }
#pragma unroll
    for (int r = 0; r < 8; ++r) {
        int m = hf * 8 + r;
        gen[(long)m * CVG + ncol]        = acc0[r];
        gen[(long)(16 + m) * CVG + ncol] = acc1[r];
    }
}

__global__ void k_gen_softmax(float* __restrict__ gen) {
    int b = blockIdx.x, i = threadIdx.x;   // blockDim = 1024
    __shared__ float sh[1024];
    float m = -1e30f;
    for (int v = i; v < CVG; v += 1024) m = fmaxf(m, gen[b * CVG + v]);
    sh[i] = m;
    __syncthreads();
    for (int s = 512; s > 0; s >>= 1) {
        if (i < s) sh[i] = fmaxf(sh[i], sh[i + s]);
        __syncthreads();
    }
    m = sh[0];
    __syncthreads();
    float sum = 0.f;
    for (int v = i; v < CVG; v += 1024) {
        float e = __expf(gen[b * CVG + v] - m);
        gen[b * CVG + v] = e;
        sum += e;
    }
    sh[i] = sum;
    __syncthreads();
    for (int s = 512; s > 0; s >>= 1) {
        if (i < s) sh[i] += sh[i + s];
        __syncthreads();
    }
    float inv = 1.f / sh[0];
    for (int v = i; v < CVG; v += 1024) gen[b * CVG + v] *= inv;
}

__global__ void k_zero_step(float* __restrict__ copybuf, float* __restrict__ osum,
                            unsigned long long* __restrict__ okey) {
    long i = blockIdx.x * (long)blockDim.x + threadIdx.x;
    long n = (long)CB * CV;
    long stride = (long)gridDim.x * blockDim.x;
    for (; i < n; i += stride) copybuf[i] = 0.f;
    int t = blockIdx.x * blockDim.x + threadIdx.x;
    if (t < CB) { osum[t] = 0.f; okey[t] = 0ull; }
}

__global__ void k_scatter(const int* __restrict__ p_batch, const float* __restrict__ attn,
                          float* __restrict__ copybuf) {
    int i = blockIdx.x * blockDim.x + threadIdx.x;
    if (i >= CB * CTP) return;
    int b = i / CTP, t = i % CTP;
    int vid = p_batch[t * CB + b];
    atomicAdd(&copybuf[(long)b * CV + vid], attn[b * CTP + t]);
}

__global__ void k_out1(const float* __restrict__ gen, const float* __restrict__ copybuf,
                       const float* __restrict__ pgen, float* __restrict__ osum,
                       unsigned long long* __restrict__ okey) {
    int b = blockIdx.y;
    int v = blockIdx.x * blockDim.x + threadIdx.x;
    float pg = pgen[b];
    float val = 0.f;
    unsigned long long key = 0ull;
    if (v < CV) {
        float g = (v < CVG) ? gen[b * CVG + v] * pg : 0.f;
        val = fmaxf(g + (1.f - pg) * copybuf[(long)b * CV + v], 1e-12f);
        key = ((unsigned long long)__float_as_uint(val) << 32) |
              (unsigned long long)(0xFFFFFFFFu - (unsigned)v);   // first-max tiebreak
    }
    __shared__ float shs[512];
    __shared__ unsigned long long shk[512];
    int i = threadIdx.x;
    shs[i] = (v < CV) ? val : 0.f;
    shk[i] = key;
    __syncthreads();
    for (int s = 256; s > 0; s >>= 1) {
        if (i < s) {
            shs[i] += shs[i + s];
            shk[i] = (shk[i] > shk[i + s]) ? shk[i] : shk[i + s];
        }
        __syncthreads();
    }
    if (i == 0) {
        atomicAdd(&osum[b], shs[0]);
        atomicMax(&okey[b], shk[0]);
    }
}

__global__ void k_out2(const float* __restrict__ gen, const float* __restrict__ copybuf,
                       const float* __restrict__ pgen, const float* __restrict__ osum,
                       float* __restrict__ out_step) {
    int b = blockIdx.y;
    int v = blockIdx.x * blockDim.x + threadIdx.x;
    if (v >= CV) return;
    float pg = pgen[b];
    float g = (v < CVG) ? gen[b * CVG + v] * pg : 0.f;
    float val = fmaxf(g + (1.f - pg) * copybuf[(long)b * CV + v], 1e-12f);
    out_step[(long)b * CV + v] = __logf(val) - __logf(osum[b]);
}

__global__ void k_feedback(const unsigned long long* __restrict__ okey, int* __restrict__ tok) {
    int b = threadIdx.x;
    if (b < CB) tok[b] = (int)(0xFFFFFFFFu - (unsigned)(okey[b] & 0xFFFFFFFFull));
}

// ---------------- host launcher ----------------

extern "C" void kernel_launch(void* const* d_in, const int* in_sizes, int n_in,
                              void* d_out, int out_size, void* d_ws, size_t ws_size,
                              hipStream_t stream) {
    const int*   p_batch   = (const int*)d_in[0];
    const int*   a_batch   = (const int*)d_in[1];
    const float* enc_hidden= (const float*)d_in[2];
    const float* context   = (const float*)d_in[3];
    const float* word_lut  = (const float*)d_in[4];
    const float* W_ih      = (const float*)d_in[5];
    const float* W_hh      = (const float*)d_in[6];
    const float* b_ih      = (const float*)d_in[7];
    const float* b_hh      = (const float*)d_in[8];
    const float* W_h       = (const float*)d_in[9];
    const float* W_s       = (const float*)d_in[10];
    const float* w_c       = (const float*)d_in[11];
    const float* v_attn    = (const float*)d_in[12];
    const float* b_attn    = (const float*)d_in[13];
    const float* W_out     = (const float*)d_in[14];
    const float* b_out     = (const float*)d_in[15];
    const float* W_gen     = (const float*)d_in[16];
    const float* b_gen     = (const float*)d_in[17];
    const float* w_p1      = (const float*)d_in[18];
    const float* w_p2      = (const float*)d_in[19];
    float* out   = (float*)d_out;
    float* trace = out + (size_t)CTDEC * CB * CV;   // p_gens (TDEC,)

    char* base = (char*)d_ws;
    size_t off = 0;
    auto carve = [&](size_t bytes) -> void* {
        off = (off + 255) & ~(size_t)255;
        void* r = base + off;
        off += bytes;
        return r;
    };
    __bf16* wgen_bf  = (__bf16*)carve((size_t)CVG * CH * 2);
    __bf16* wx_bf    = (__bf16*)carve((size_t)GATES * KLSTM * 2);
    float*  ctx_proj = (float*)carve((size_t)CTP * CB * CH * 4);
    float*  h_buf    = (float*)carve((size_t)CB * CH * 4);
    float*  c_buf    = (float*)carve((size_t)CB * CH * 4);
    float*  cov      = (float*)carve((size_t)CB * CTP * 4);
    float*  emb      = (float*)carve((size_t)CB * CE * 4);
    __bf16* x_bf     = (__bf16*)carve((size_t)CB * KLSTM * 2);
    float*  gates    = (float*)carve((size_t)CB * GATES * 4);
    float*  sproj    = (float*)carve((size_t)CB * CH * 4);
    float*  scores   = (float*)carve((size_t)CB * CTP * 4);
    float*  attn     = (float*)carve((size_t)CB * CTP * 4);
    float*  ctxvec   = (float*)carve((size_t)CB * CH * 4);
    float*  attnh    = (float*)carve((size_t)CB * CH * 4);
    __bf16* attnh_bf = (__bf16*)carve((size_t)CB * CH * 2);
    float*  pgen     = (float*)carve((size_t)CB * 4);
    float*  gen      = (float*)carve((size_t)CB * CVG * 4);
    float*  copybuf  = (float*)carve((size_t)CB * CV * 4);
    float*  osum     = (float*)carve((size_t)CB * 4);
    unsigned long long* okey = (unsigned long long*)carve((size_t)CB * 8);
    int*    tok      = (int*)carve((size_t)CB * 4);

    // one-time prep (deterministic, re-done every launch)
    k_cvt_wgen<<<(CVG * CH + 255) / 256, 256, 0, stream>>>(W_gen, wgen_bf);
    k_build_wx<<<(GATES * KLSTM + 255) / 256, 256, 0, stream>>>(W_ih, W_hh, wx_bf);
    k_ctxproj<<<CTP * CB, CH, 0, stream>>>(context, W_h, ctx_proj);
    k_init<<<(CB * CTP + 255) / 256, 256, 0, stream>>>(enc_hidden, a_batch, h_buf, c_buf, cov, tok);

    const int genNT     = CVG / 16;                      // 1250 N-tiles (1 wave each)
    const int genBlocks = (genNT + 7) / 8;               // 157 blocks of 8 waves
    const int gatesNT   = GATES / 16;                    // 64
    const int gatesBlocks = (gatesNT + 7) / 8;           // 8
    const dim3 vgrid((CV + 511) / 512, CB);

    for (int t = 0; t < CTDEC; ++t) {
        k_stage_x<<<CB, KLSTM, 0, stream>>>(word_lut, tok, h_buf, emb, x_bf);
        k_gates_wmma<<<gatesBlocks, 256, 0, stream>>>(x_bf, wx_bf, b_ih, b_hh, gates);
        k_lstm<<<CB, CH, 0, stream>>>(gates, h_buf, c_buf);
        k_sproj<<<CB, CH, 0, stream>>>(h_buf, W_s, sproj);
        k_score<<<dim3(CTP, CB), CH, 0, stream>>>(ctx_proj, sproj, cov, w_c, b_attn, v_attn, scores);
        k_softmax_attn<<<CB, 512, 0, stream>>>(scores, attn, cov);
        k_ctxvec<<<CB, CH, 0, stream>>>(attn, context, ctxvec);
        k_attnh<<<CB, CH, 0, stream>>>(ctxvec, h_buf, W_out, b_out, attnh, attnh_bf);
        k_pgen<<<CB, CH, 0, stream>>>(attnh, emb, w_p1, w_p2, pgen, trace, t);
        k_gen_wmma<<<genBlocks, 256, 0, stream>>>(attnh_bf, wgen_bf, b_gen, gen);
        k_gen_softmax<<<CB, 1024, 0, stream>>>(gen);
        k_zero_step<<<512, 256, 0, stream>>>(copybuf, osum, okey);
        k_scatter<<<(CB * CTP + 255) / 256, 256, 0, stream>>>(p_batch, attn, copybuf);
        k_out1<<<vgrid, 512, 0, stream>>>(gen, copybuf, pgen, osum, okey);
        k_out2<<<vgrid, 512, 0, stream>>>(gen, copybuf, pgen, osum,
                                          out + (size_t)t * CB * CV);
        k_feedback<<<1, CB, 0, stream>>>(okey, tok);
    }
}
